// MPNNLayer_84696755077493
// MI455X (gfx1250) — matrix operand order repaired and checked by reading, not compiled, not run on hardware
//
#include <hip/hip_runtime.h>
#include <hip/hip_bf16.h>

typedef __attribute__((ext_vector_type(16))) _Float16 v16h;
typedef __attribute__((ext_vector_type(8)))  _Float16 v8h;
typedef __attribute__((ext_vector_type(8)))  float    v8f;

#define EPSF 1e-8f

// ---- packed-weight offsets (in halves); each matrix padded K->mult(32), N->mult(16),
// fragment = 512 halves laid out as frag[lane(N=lane&15)][j(K=8*(lane>>4)+(j&7)+16*(j>>3))] ----
#define OFF_WS1 0        // 232x100 -> 8x7 frags
#define OFF_WS2 28672    // 116x100 -> 4x7
#define OFF_WS3 43008    // 116x100 -> 4x7
#define OFF_WS4 57344    // 132x400 -> 5x25
#define OFF_WS5 121344   // 432x100 -> 14x7
#define OFF_WH1 171520   // 32x32 -> 1x2
#define OFF_WH2 172544   // 16x16 -> 1x1
#define OFF_WH3 173056
#define OFF_WH4 173568   // 16x32 -> 1x2
#define OFF_WH5 174592   // 32x32 -> 1x2
#define OFF_WV1 175616   // 32x16 -> 1x1
#define OFF_WV2 176128   // 16x16 -> 1x1
#define OFF_WV3 176640
#define OFF_WV4 177152   // 32x32 -> 1x2
#define OFF_WV5 178176   // 32x16 -> 1x1
#define PW_TOTAL 178688

__device__ __forceinline__ v8f zero8() {
    v8f z;
#pragma unroll
    for (int i = 0; i < 8; ++i) z[i] = 0.f;
    return z;
}

__device__ __forceinline__ v8f wmma32(v16h a, v16h b, v8f c) {
    return __builtin_amdgcn_wmma_f32_16x16x32_f16(false, a, false, b, (short)0, c, false, false);
}

__device__ __forceinline__ float sigm(float x) {
    return 1.f / (1.f + __expf(-x));
}

// A fragment (16xK f16, row-major in LDS, ldk halves): per ISA 16-bit A layout.
// lane<16:  M=lane, halves = K[kc+0..7], K[kc+16..23]
// lane>=16: M=lane-16, halves = K[kc+8..15], K[kc+24..31]
__device__ __forceinline__ v16h lds_load_a(const _Float16* A, int ldk, int kc) {
    int lane = threadIdx.x & 31;
    int m    = lane & 15;
    int khi  = (lane >> 4) ? 8 : 0;
    const _Float16* r = A + m * ldk + kc + khi;
    v8h lo = *(const v8h*)(r);
    v8h hi = *(const v8h*)(r + 16);
    return __builtin_shufflevector(lo, hi, 0,1,2,3,4,5,6,7,8,9,10,11,12,13,14,15);
}

// B fragment from packed weights: frag base + lane*16 halves (two 16B loads)
__device__ __forceinline__ v16h ldg_b(const _Float16* P, int frag) {
    int lane = threadIdx.x & 31;
    const _Float16* p = P + ((size_t)frag << 9) + (lane << 4);
    v8h lo = *(const v8h*)(p);
    v8h hi = *(const v8h*)(p + 8);
    return __builtin_shufflevector(lo, hi, 0,1,2,3,4,5,6,7,8,9,10,11,12,13,14,15);
}

// C/D tile (16x16 f32): lane holds N = n0+(lane&15), rows M = ((lane>>4)<<3)+r
__device__ __forceinline__ void st_c_f16(v8f c, _Float16* D, int ldn, int n0, int ncap) {
    int lane = threadIdx.x & 31;
    int n = n0 + (lane & 15);
    if (n >= ncap) return;
    int mb = (lane >> 4) << 3;
#pragma unroll
    for (int r = 0; r < 8; ++r) D[(mb + r) * ldn + n] = (_Float16)c[r];
}

__device__ __forceinline__ void st_c_f32(v8f c, float* D, int ldn, int n0, int ncap) {
    int lane = threadIdx.x & 31;
    int n = n0 + (lane & 15);
    if (n >= ncap) return;
    int mb = (lane >> 4) << 3;
#pragma unroll
    for (int r = 0; r < 8; ++r) D[(mb + r) * ldn + n] = c[r];
}

// ---------------- weight pre-pack: W (KxN f32 row-major) -> f16 WMMA-B fragments ----------
__global__ void pack_kernel(const float* __restrict__ W, int K, int N, int Kc, int Nt,
                            _Float16* __restrict__ dst) {
    int i = blockIdx.x * blockDim.x + threadIdx.x;
    int total = Kc * Nt * 512;
    if (i >= total) return;
    int frag = i >> 9;
    int r    = i & 511;
    int lane = r >> 4;      // N within 16-tile = lane&15, K half = lane>>4
    int j    = r & 15;      // K = 8*(lane>>4) + (j&7) + 16*(j>>3) within 32-chunk
    int kcI  = frag / Nt, nt = frag % Nt;
    int k = kcI * 32 + ((lane >> 4) << 3) + (j & 7) + ((j >> 3) << 4), n = nt * 16 + (lane & 15);
    float v = (k < K && n < N) ? W[k * N + n] : 0.f;
    dst[i] = (_Float16)v;
}

// ---------------- edge kernel: gvp1..gvp3 + masked mean over K=30 -> dh ----------------
__global__ __launch_bounds__(64)
void edge_kernel(const float* __restrict__ hV, const float* __restrict__ hM,
                 const int* __restrict__ maskA, const _Float16* __restrict__ pw,
                 const float* __restrict__ bs1, const float* __restrict__ bs2,
                 const float* __restrict__ bs3, float* __restrict__ dh) {
    __shared__ __align__(16) _Float16 sA [2][16][256];   // gvp1 scalar A: [sV|sM|vn1|0]
    __shared__ __align__(16) _Float16 vA [2][3][16][32]; // gvp1 vector A per component
    __shared__ __align__(16) _Float16 vhA[2][3][16][32]; // vh staging per gvp
    __shared__ __align__(16) _Float16 sB2[2][16][128];   // gvp2 scalar A: [s1|vn2|0]
    __shared__ __align__(16) _Float16 vB2[2][3][16][32]; // gvp2/3 vector A (vo outputs)
    __shared__ float dhred[148];
    __shared__ float hVs[148];
    __shared__ int   am[32];

    const int node = blockIdx.x;
    const int tid = threadIdx.x;
    const int w = tid >> 5, lane = tid & 31;

    for (int i = tid; i < 148; i += 64) { hVs[i] = hV[(size_t)node * 148 + i]; dhred[i] = 0.f; }
    for (int i = tid; i < 32; i += 64) am[i] = (i < 30) ? maskA[node * 30 + i] : 0;
    if (lane < 16) {
        for (int j = 232; j < 256; ++j) sA[w][lane][j] = (_Float16)0.f;
        for (int j = 116; j < 128; ++j) sB2[w][lane][j] = (_Float16)0.f;
        for (int c = 0; c < 3; ++c)
            for (int j = 16; j < 32; ++j) vB2[w][c][lane][j] = (_Float16)0.f;
    }
    __syncthreads();

    // stage this wave's 16 edges (k = w*16+e); invalid edges get zero message input
    for (int e = 0; e < 16; ++e) {
        int k = w * 16 + e;
        bool val = (k < 30);
        const float* row = hM + ((size_t)node * 30 + (val ? k : 0)) * 148;
        __builtin_prefetch(row, 0, 1);
        for (int i = lane; i < 148; i += 32) {
            float x = val ? row[i] : 0.f;
            if (i < 48) vA[w][i % 3][e][16 + i / 3] = (_Float16)x;
            else        sA[w][e][100 + (i - 48)]    = (_Float16)x;
            float y = hVs[i];
            if (i < 48) vA[w][i % 3][e][i / 3] = (_Float16)y;
            else        sA[w][e][i - 48]       = (_Float16)y;
        }
    }
    __syncthreads();

    float rm[8]; // per-lane row masks (row = edge index)
    {
        int mb = (lane >> 4) << 3;
#pragma unroll
        for (int r = 0; r < 8; ++r) rm[r] = (float)am[w * 16 + mb + r];
    }

    // ---- GVP1 ---- vh1 = [vV|vM] @ Wh1 (16x32x32 per component)
    {
        v8f ss[2] = { zero8(), zero8() };
        for (int c = 0; c < 3; ++c) {
            v16h a = lds_load_a(&vA[w][c][0][0], 32, 0);
            for (int nt = 0; nt < 2; ++nt) {
                v8f acc = wmma32(a, ldg_b(pw + OFF_WH1, nt), zero8());
#pragma unroll
                for (int i = 0; i < 8; ++i) ss[nt][i] += acc[i] * acc[i];
                st_c_f16(acc, &vhA[w][c][0][0], 32, nt * 16, 32);
            }
        }
        for (int nt = 0; nt < 2; ++nt) {
            v8f vn;
#pragma unroll
            for (int i = 0; i < 8; ++i) vn[i] = sqrtf(ss[nt][i] + EPSF);
            st_c_f16(vn, &sA[w][0][0], 256, 200 + nt * 16, 232);
        }
    }
    __syncthreads();
    // vo1 = vh1 @ Wv1, gated by sigmoid(|vo1|)
    {
        v8f av[3]; v8f ss = zero8();
        for (int c = 0; c < 3; ++c) {
            v16h a = lds_load_a(&vhA[w][c][0][0], 32, 0);
            av[c] = wmma32(a, ldg_b(pw + OFF_WV1, 0), zero8());
#pragma unroll
            for (int i = 0; i < 8; ++i) ss[i] += av[c][i] * av[c][i];
        }
        v8f g;
#pragma unroll
        for (int i = 0; i < 8; ++i) g[i] = sigm(sqrtf(ss[i] + EPSF));
        for (int c = 0; c < 3; ++c) {
            v8f o;
#pragma unroll
            for (int i = 0; i < 8; ++i) o[i] = av[c][i] * g[i];
            st_c_f16(o, &vB2[w][c][0][0], 32, 0, 16);
        }
        if (lane < 16)  // vh2/vh3 are only 16 wide: zero upper K half of vhA once
            for (int c = 0; c < 3; ++c)
                for (int j = 16; j < 32; ++j) vhA[w][c][lane][j] = (_Float16)0.f;
    }
    __syncthreads();
    // s1 = [sV|sM|vn1] @ Ws1 + bs1, relu -> sB2[:, :100]
    {
        v16h a1[8];
#pragma unroll
        for (int kc = 0; kc < 8; ++kc) a1[kc] = lds_load_a(&sA[w][0][0], 256, kc * 32);
        for (int nt = 0; nt < 7; ++nt) {
            v8f acc = zero8();
#pragma unroll
            for (int kc = 0; kc < 8; ++kc)
                acc = wmma32(a1[kc], ldg_b(pw + OFF_WS1, kc * 7 + nt), acc);
            int n = nt * 16 + (lane & 15);
            float b = (n < 100) ? bs1[n] : 0.f;
#pragma unroll
            for (int i = 0; i < 8; ++i) acc[i] = fmaxf(acc[i] + b, 0.f);
            st_c_f16(acc, &sB2[w][0][0], 128, nt * 16, 100);
        }
    }
    __syncthreads();

    // ---- GVP2 ---- vh2 = vo1 @ Wh2
    {
        v8f ss = zero8();
        for (int c = 0; c < 3; ++c) {
            v16h a = lds_load_a(&vB2[w][c][0][0], 32, 0);
            v8f acc = wmma32(a, ldg_b(pw + OFF_WH2, 0), zero8());
#pragma unroll
            for (int i = 0; i < 8; ++i) ss[i] += acc[i] * acc[i];
            st_c_f16(acc, &vhA[w][c][0][0], 32, 0, 16);
        }
        v8f vn;
#pragma unroll
        for (int i = 0; i < 8; ++i) vn[i] = sqrtf(ss[i] + EPSF);
        st_c_f16(vn, &sB2[w][0][0], 128, 100, 116);
    }
    __syncthreads();
    // vo2 = vh2 @ Wv2 (gated) -> vB2
    {
        v8f av[3]; v8f ss = zero8();
        for (int c = 0; c < 3; ++c) {
            v16h a = lds_load_a(&vhA[w][c][0][0], 32, 0);
            av[c] = wmma32(a, ldg_b(pw + OFF_WV2, 0), zero8());
#pragma unroll
            for (int i = 0; i < 8; ++i) ss[i] += av[c][i] * av[c][i];
        }
        v8f g;
#pragma unroll
        for (int i = 0; i < 8; ++i) g[i] = sigm(sqrtf(ss[i] + EPSF));
        for (int c = 0; c < 3; ++c) {
            v8f o;
#pragma unroll
            for (int i = 0; i < 8; ++i) o[i] = av[c][i] * g[i];
            st_c_f16(o, &vB2[w][c][0][0], 32, 0, 16);
        }
    }
    __syncthreads();
    // s2 = [s1|vn2] @ Ws2 + bs2, relu -> sA[:, :100]
    {
        v16h a2[4];
#pragma unroll
        for (int kc = 0; kc < 4; ++kc) a2[kc] = lds_load_a(&sB2[w][0][0], 128, kc * 32);
        for (int nt = 0; nt < 7; ++nt) {
            v8f acc = zero8();
#pragma unroll
            for (int kc = 0; kc < 4; ++kc)
                acc = wmma32(a2[kc], ldg_b(pw + OFF_WS2, kc * 7 + nt), acc);
            int n = nt * 16 + (lane & 15);
            float b = (n < 100) ? bs2[n] : 0.f;
#pragma unroll
            for (int i = 0; i < 8; ++i) acc[i] = fmaxf(acc[i] + b, 0.f);
            st_c_f16(acc, &sA[w][0][0], 256, nt * 16, 100);
        }
    }
    if (lane < 16)
        for (int j = 116; j < 128; ++j) sA[w][lane][j] = (_Float16)0.f;
    __syncthreads();

    // ---- GVP3 (no nonlinearities) ---- vh3 = vo2 @ Wh3
    {
        v8f ss = zero8();
        for (int c = 0; c < 3; ++c) {
            v16h a = lds_load_a(&vB2[w][c][0][0], 32, 0);
            v8f acc = wmma32(a, ldg_b(pw + OFF_WH3, 0), zero8());
#pragma unroll
            for (int i = 0; i < 8; ++i) ss[i] += acc[i] * acc[i];
            st_c_f16(acc, &vhA[w][c][0][0], 32, 0, 16);
        }
        v8f vn;
#pragma unroll
        for (int i = 0; i < 8; ++i) vn[i] = sqrtf(ss[i] + EPSF);
        st_c_f16(vn, &sA[w][0][0], 256, 100, 116);
    }
    __syncthreads();
    // vo3 -> masked row-sum into dhred (vector features o*3+c)
    for (int c = 0; c < 3; ++c) {
        v16h a = lds_load_a(&vhA[w][c][0][0], 32, 0);
        v8f acc = wmma32(a, ldg_b(pw + OFF_WV3, 0), zero8());
        float s = 0.f;
#pragma unroll
        for (int r = 0; r < 8; ++r) s += acc[r] * rm[r];
        s += __shfl_down(s, 16, 32);
        if (lane < 16) atomicAdd(&dhred[lane * 3 + c], s);
    }
    // s3 = [s2|vn3] @ Ws3 + bs3 -> masked row-sum into dhred (scalar features)
    {
        v16h a3[4];
#pragma unroll
        for (int kc = 0; kc < 4; ++kc) a3[kc] = lds_load_a(&sA[w][0][0], 256, kc * 32);
        for (int nt = 0; nt < 7; ++nt) {
            v8f acc = zero8();
#pragma unroll
            for (int kc = 0; kc < 4; ++kc)
                acc = wmma32(a3[kc], ldg_b(pw + OFF_WS3, kc * 7 + nt), acc);
            int n = nt * 16 + (lane & 15);
            float b = (n < 100) ? bs3[n] : 0.f;
            float s = 0.f;
#pragma unroll
            for (int r = 0; r < 8; ++r) s += (acc[r] + b) * rm[r];
            s += __shfl_down(s, 16, 32);
            if (lane < 16 && n < 100) atomicAdd(&dhred[48 + n], s);
        }
    }
    __syncthreads();
    for (int i = tid; i < 148; i += 64) dh[(size_t)node * 148 + i] = dhred[i] * (1.f / 30.f);
}

// ---------------- node kernel: LN0, gvp4, gvp5, LN1, mask; 16 nodes per wave ------------
__global__ __launch_bounds__(32)
void node_kernel(const float* __restrict__ hV, const float* __restrict__ dh,
                 const int* __restrict__ maskV, const _Float16* __restrict__ pw,
                 const float* __restrict__ bs4, const float* __restrict__ bs5,
                 const float* __restrict__ g0, const float* __restrict__ b0,
                 const float* __restrict__ g1, const float* __restrict__ b1,
                 float* __restrict__ out) {
    __shared__ float hn[16][148];                       // normalized h (residual source)
    __shared__ __align__(16) _Float16 sA4[16][160];     // [s|vn4|0]
    __shared__ __align__(16) _Float16 vA4[3][16][32];
    __shared__ __align__(16) _Float16 vhA[3][16][32];   // vh4 then vh5
    __shared__ __align__(16) _Float16 sA5[16][448];     // [s4|vn5|0]
    __shared__ __align__(16) _Float16 vA5[3][16][32];
    __shared__ float d2s[16][112];
    __shared__ float d2v[3][16][16];

    const int lane = threadIdx.x;
    const int base = blockIdx.x * 16;

    if (lane < 16) {
        for (int j = 132; j < 160; ++j) sA4[lane][j] = (_Float16)0.f;
        for (int j = 432; j < 448; ++j) sA5[lane][j] = (_Float16)0.f;
        for (int c = 0; c < 3; ++c)
            for (int j = 16; j < 32; ++j) vA4[c][lane][j] = (_Float16)0.f;
    }
    // LN0 per node (one lane per node)
    if (lane < 16) {
        const float* ph = hV + (size_t)(base + lane) * 148;
        const float* pd = dh + (size_t)(base + lane) * 148;
        for (int i = 0; i < 148; ++i) hn[lane][i] = ph[i] + pd[i];
        float vs = 0.f;
        for (int i = 0; i < 48; ++i) { float t = hn[lane][i]; vs += t * t; }
        float den = sqrtf(vs * (1.f / 16.f) + EPSF);
        float mu = 0.f;
        for (int i = 48; i < 148; ++i) mu += hn[lane][i];
        mu *= 0.01f;
        float var = 0.f;
        for (int i = 48; i < 148; ++i) { float d = hn[lane][i] - mu; var += d * d; }
        var *= 0.01f;
        float inv = 1.f / sqrtf(var + 1e-5f);
        for (int i = 0; i < 48; ++i) {
            float t = hn[lane][i] / den; hn[lane][i] = t;
            vA4[i % 3][lane][i / 3] = (_Float16)t;
        }
        for (int i = 48; i < 148; ++i) {
            int j = i - 48;
            float t = (hn[lane][i] - mu) * inv * g0[j] + b0[j];
            hn[lane][i] = t;
            sA4[lane][j] = (_Float16)t;
        }
    }
    __syncthreads();

    // ---- GVP4 ---- vh4 = v @ Wh4 (16->32)
    {
        v8f ss[2] = { zero8(), zero8() };
        for (int c = 0; c < 3; ++c) {
            v16h a = lds_load_a(&vA4[c][0][0], 32, 0);
            for (int nt = 0; nt < 2; ++nt) {
                v8f acc = wmma32(a, ldg_b(pw + OFF_WH4, nt), zero8());
#pragma unroll
                for (int i = 0; i < 8; ++i) ss[nt][i] += acc[i] * acc[i];
                st_c_f16(acc, &vhA[c][0][0], 32, nt * 16, 32);
            }
        }
        for (int nt = 0; nt < 2; ++nt) {
            v8f vn;
#pragma unroll
            for (int i = 0; i < 8; ++i) vn[i] = sqrtf(ss[nt][i] + EPSF);
            st_c_f16(vn, &sA4[0][0], 160, 100 + nt * 16, 132);
        }
    }
    __syncthreads();
    // vo4 = vh4 @ Wv4 (gated) -> vA5 (gvp5 vector input, 32 wide)
    {
        v8f av[3][2]; v8f ss[2] = { zero8(), zero8() };
        for (int c = 0; c < 3; ++c) {
            v16h a = lds_load_a(&vhA[c][0][0], 32, 0);
            for (int nt = 0; nt < 2; ++nt) {
                av[c][nt] = wmma32(a, ldg_b(pw + OFF_WV4, nt), zero8());
#pragma unroll
                for (int i = 0; i < 8; ++i) ss[nt][i] += av[c][nt][i] * av[c][nt][i];
            }
        }
        for (int nt = 0; nt < 2; ++nt) {
            v8f g;
#pragma unroll
            for (int i = 0; i < 8; ++i) g[i] = sigm(sqrtf(ss[nt][i] + EPSF));
            for (int c = 0; c < 3; ++c) {
                v8f o;
#pragma unroll
                for (int i = 0; i < 8; ++i) o[i] = av[c][nt][i] * g[i];
                st_c_f16(o, &vA5[c][0][0], 32, nt * 16, 32);
            }
        }
    }
    __syncthreads();
    // s4 = [s|vn4] @ Ws4 + bs4, relu -> sA5[:, :400]
    {
        v16h a4[5];
#pragma unroll
        for (int kc = 0; kc < 5; ++kc) a4[kc] = lds_load_a(&sA4[0][0], 160, kc * 32);
        for (int nt = 0; nt < 25; ++nt) {
            v8f acc = zero8();
#pragma unroll
            for (int kc = 0; kc < 5; ++kc)
                acc = wmma32(a4[kc], ldg_b(pw + OFF_WS4, kc * 25 + nt), acc);
            int n = nt * 16 + (lane & 15);
            float b = bs4[n];
#pragma unroll
            for (int i = 0; i < 8; ++i) acc[i] = fmaxf(acc[i] + b, 0.f);
            st_c_f16(acc, &sA5[0][0], 448, nt * 16, 400);
        }
    }
    __syncthreads();

    // ---- GVP5 ---- vh5 = vo4 @ Wh5
    {
        v8f ss[2] = { zero8(), zero8() };
        for (int c = 0; c < 3; ++c) {
            v16h a = lds_load_a(&vA5[c][0][0], 32, 0);
            for (int nt = 0; nt < 2; ++nt) {
                v8f acc = wmma32(a, ldg_b(pw + OFF_WH5, nt), zero8());
#pragma unroll
                for (int i = 0; i < 8; ++i) ss[nt][i] += acc[i] * acc[i];
                st_c_f16(acc, &vhA[c][0][0], 32, nt * 16, 32);
            }
        }
        for (int nt = 0; nt < 2; ++nt) {
            v8f vn;
#pragma unroll
            for (int i = 0; i < 8; ++i) vn[i] = sqrtf(ss[nt][i] + EPSF);
            st_c_f16(vn, &sA5[0][0], 448, 400 + nt * 16, 432);
        }
    }
    __syncthreads();
    // vo5 = vh5 @ Wv5 (no gate) -> d2v
    for (int c = 0; c < 3; ++c) {
        v16h a = lds_load_a(&vhA[c][0][0], 32, 0);
        v8f acc = wmma32(a, ldg_b(pw + OFF_WV5, 0), zero8());
        st_c_f32(acc, &d2v[c][0][0], 16, 0, 16);
    }
    // s5 = [s4|vn5] @ Ws5 + bs5 (no relu) -> d2s
    {
        v16h a5[14];
#pragma unroll
        for (int kc = 0; kc < 14; ++kc) a5[kc] = lds_load_a(&sA5[0][0], 448, kc * 32);
        for (int nt = 0; nt < 7; ++nt) {
            v8f acc = zero8();
#pragma unroll
            for (int kc = 0; kc < 14; ++kc)
                acc = wmma32(a5[kc], ldg_b(pw + OFF_WS5, kc * 7 + nt), acc);
            int n = nt * 16 + (lane & 15);
            float b = (n < 100) ? bs5[n] : 0.f;
#pragma unroll
            for (int i = 0; i < 8; ++i) acc[i] += b;
            st_c_f32(acc, &d2s[0][0], 112, nt * 16, 112);
        }
    }
    __syncthreads();

    // residual + LN1 + mask + store
    if (lane < 16) {
        int node = base + lane;
        for (int i = 0; i < 48; ++i)  hn[lane][i] += d2v[i % 3][lane][i / 3];
        for (int i = 48; i < 148; ++i) hn[lane][i] += d2s[lane][i - 48];
        float vs = 0.f;
        for (int i = 0; i < 48; ++i) { float t = hn[lane][i]; vs += t * t; }
        float den = sqrtf(vs * (1.f / 16.f) + EPSF);
        float mu = 0.f;
        for (int i = 48; i < 148; ++i) mu += hn[lane][i];
        mu *= 0.01f;
        float var = 0.f;
        for (int i = 48; i < 148; ++i) { float d = hn[lane][i] - mu; var += d * d; }
        var *= 0.01f;
        float inv = 1.f / sqrtf(var + 1e-5f);
        float mv = (float)maskV[node];
        float* po = out + (size_t)node * 148;
        for (int i = 0; i < 48; ++i) po[i] = (hn[lane][i] / den) * mv;
        for (int i = 48; i < 148; ++i) {
            int j = i - 48;
            po[i] = (((hn[lane][i] - mu) * inv) * g1[j] + b1[j]) * mv;
        }
    }
}

extern "C" void kernel_launch(void* const* d_in, const int* in_sizes, int n_in,
                              void* d_out, int out_size, void* d_ws, size_t ws_size,
                              hipStream_t stream) {
    (void)in_sizes; (void)n_in; (void)out_size; (void)ws_size;
    const float* hV    = (const float*)d_in[0];
    const float* hM    = (const float*)d_in[1];
    const int*   maskV = (const int*)d_in[2];
    const int*   maskA = (const int*)d_in[3];
    const float* Wh1 = (const float*)d_in[4];  const float* Wv1 = (const float*)d_in[5];
    const float* Ws1 = (const float*)d_in[6];  const float* bs1 = (const float*)d_in[7];
    const float* Wh2 = (const float*)d_in[8];  const float* Wv2 = (const float*)d_in[9];
    const float* Ws2 = (const float*)d_in[10]; const float* bs2 = (const float*)d_in[11];
    const float* Wh3 = (const float*)d_in[12]; const float* Wv3 = (const float*)d_in[13];
    const float* Ws3 = (const float*)d_in[14]; const float* bs3 = (const float*)d_in[15];
    const float* Wh4 = (const float*)d_in[16]; const float* Wv4 = (const float*)d_in[17];
    const float* Ws4 = (const float*)d_in[18]; const float* bs4 = (const float*)d_in[19];
    const float* Wh5 = (const float*)d_in[20]; const float* Wv5 = (const float*)d_in[21];
    const float* Ws5 = (const float*)d_in[22]; const float* bs5 = (const float*)d_in[23];
    const float* g0  = (const float*)d_in[24]; const float* b0  = (const float*)d_in[25];
    const float* g1  = (const float*)d_in[26]; const float* b1  = (const float*)d_in[27];

    _Float16* pw = (_Float16*)d_ws;
    float* dh = (float*)((char*)d_ws + (512 << 10)); // dh scratch after 512KB of packed weights

    auto pack = [&](const float* W, int K, int N, int Kc, int Nt, int off) {
        int total = Kc * Nt * 512;
        pack_kernel<<<(total + 255) / 256, 256, 0, stream>>>(W, K, N, Kc, Nt, pw + off);
    };
    pack(Ws1, 232, 100, 8, 7,  OFF_WS1);
    pack(Ws2, 116, 100, 4, 7,  OFF_WS2);
    pack(Ws3, 116, 100, 4, 7,  OFF_WS3);
    pack(Ws4, 132, 400, 5, 25, OFF_WS4);
    pack(Ws5, 432, 100, 14, 7, OFF_WS5);
    pack(Wh1, 32, 32, 1, 2, OFF_WH1);
    pack(Wh2, 16, 16, 1, 1, OFF_WH2);
    pack(Wh3, 16, 16, 1, 1, OFF_WH3);
    pack(Wh4, 16, 32, 1, 2, OFF_WH4);
    pack(Wh5, 32, 32, 1, 2, OFF_WH5);
    pack(Wv1, 32, 16, 1, 1, OFF_WV1);
    pack(Wv2, 16, 16, 1, 1, OFF_WV2);
    pack(Wv3, 16, 16, 1, 1, OFF_WV3);
    pack(Wv4, 32, 32, 1, 2, OFF_WV4);
    pack(Wv5, 32, 16, 1, 1, OFF_WV5);

    edge_kernel<<<16384, 64, 0, stream>>>(hV, hM, maskA, pw, bs1, bs2, bs3, dh);
    node_kernel<<<1024, 32, 0, stream>>>(hV, dh, maskV, pw, bs4, bs5, g0, b0, g1, b1,
                                         (float*)d_out);
}
